// NeuralODE_32873679684315
// MI455X (gfx1250) — matrix operand order repaired and checked by reading, not compiled, not run on hardware
//
#include <hip/hip_runtime.h>
#include <math.h>

typedef __attribute__((ext_vector_type(16))) _Float16     v16h;
typedef __attribute__((ext_vector_type(8)))  float        v8f;
typedef __attribute__((ext_vector_type(8)))  unsigned int v8u;

#define WMMA_F16(A, B, C) \
  __builtin_amdgcn_wmma_f32_16x16x32_f16(false, (A), false, (B), (short)0, (C), false, false)

__device__ __forceinline__ float fast_tanh(float x) {
#if __has_builtin(__builtin_amdgcn_tanhf)
  return __builtin_amdgcn_tanhf(x);   // native v_tanh_f32 on gfx1250
#else
  return tanhf(x);
#endif
}

// Per-wave LDS ordering: wait for all of this wave's DS ops. Cross-lane RAW
// within the wave is the only hazard (each wave has private scratch buffers).
__device__ __forceinline__ void lds_wave_fence() {
  asm volatile("s_wait_dscnt 0" ::: "memory");
}

// A-fragment (16x32, f16) from row-major f16 LDS matrix W[M][Kdim].
// ISA layout: lanes 0-15 row m: VGPR0-3 K=0..7, VGPR4-7 K=16..23;
//             lanes 16-31 row m: VGPR0-3 K=8..15, VGPR4-7 K=24..31.
__device__ __forceinline__ v16h load_a_frag(const _Float16* W, int Kdim, int m0,
                                            int k0, int lane) {
  const int row  = m0 + (lane & 15);
  const int koff = (lane >> 4) ? 8 : 0;
  v8u tmp;
#pragma unroll
  for (int v = 0; v < 8; ++v) {
    int kk  = k0 + ((v >> 2) << 4) + koff + ((v & 3) << 1);
    tmp[v] = *(const unsigned int*)(W + row * Kdim + kk);  // 2 packed f16
  }
  return __builtin_bit_cast(v16h, tmp);
}

// B-fragment (32x16, f16) from column-major f16 LDS activations act[col][k],
// column stride 66 (padded to spread LDS banks).
// ISA layout: lanes 0-15 (cols): VGPR0-3 K=0..7, VGPR4-7 K=16..23; lanes 16-31: K=8..15, K=24..31.
__device__ __forceinline__ v16h load_b_frag(const _Float16* act, int k0, int lane) {
  const int col = lane & 15;
  const int kb  = k0 + ((lane >> 4) << 3);
  const _Float16* p = act + col * 66 + kb;
  v8u tmp;
#pragma unroll
  for (int v = 0; v < 8; ++v) tmp[v] = *(const unsigned int*)(p + ((v >> 2) << 4) + ((v & 3) << 1));
  return __builtin_bit_cast(v16h, tmp);
}

struct Frags {
  v16h w2[4][2];  // pW2 (64x64): 4 M-tiles x 2 K-tiles
  v16h w3[4][2];  // pW3 (64x64)
  v16h t2[2];     // tW2 (32x32): 2 M-tiles x 1 K-tile
};

// One cooperative vector-field evaluation for a 16-element tile.
// Lanes n and n+16 are replicas of element n. phi and theta branches are
// independent until the final f*(dW.K): their layer-1 stores share one fence
// and theta's WMMAs/tanh-tail overlap phi-L2's WMMA hazard windows.
__device__ __forceinline__ float vf_eval(
    float t1, float tau_s, float y_s, const Frags& F,
    const float* sP1, const float* sPb1, const float* sPb2, const float* sPb3,
    const float* sDW, const float* sT1w, const float* sTb1, const float* sTb2,
    const float* sT3, _Float16* actA, _Float16* actB, _Float16* actG,
    float db0, float tb30, int lane) {
  const int col = lane & 15;
  const int hi  = lane >> 4;

  // ---- phi L1: h1 = tanh(pW1 @ [t1, tau] + pb1), 64 rows -> actA ----
#pragma unroll
  for (int j = 0; j < 32; ++j) {
    int   i = hi * 32 + j;
    float v = fmaf(sP1[2 * i], t1, fmaf(sP1[2 * i + 1], tau_s, sPb1[i]));
    actA[col * 66 + i] = (_Float16)fast_tanh(v);
  }
  // ---- theta L1: g1 = tanh(tW1 @ [tau, y] + tb1), 32 rows -> actG ----
#pragma unroll
  for (int j = 0; j < 16; ++j) {
    int   i = hi * 16 + j;
    float v = fmaf(sT1w[2 * i], tau_s, fmaf(sT1w[2 * i + 1], y_s, sTb1[i]));
    actG[col * 66 + i] = (_Float16)fast_tanh(v);
  }
  lds_wave_fence();

  v16h b0 = load_b_frag(actA, 0, lane);
  v16h b1 = load_b_frag(actA, 32, lane);
  v16h gb = load_b_frag(actG, 0, lane);

  // ---- theta L2 + head (independent of phi; fills phi WMMA hazard slots) ----
  float f = 0.0f;
#pragma unroll
  for (int mt = 0; mt < 2; ++mt) {
    v8f acc = {};
    acc = WMMA_F16(F.t2[mt], gb, acc);
#pragma unroll
    for (int r = 0; r < 8; ++r) {
      int i = mt * 16 + hi * 8 + r;
      f     = fmaf(sT3[i], fast_tanh(acc[r] + sTb2[i]), f);
    }
  }

  // ---- phi L2: h2 = tanh(pW2 @ h1 + pb2) -> actB ----
#pragma unroll
  for (int mt = 0; mt < 4; ++mt) {
    v8f acc = {};
    acc = WMMA_F16(F.w2[mt][0], b0, acc);
    acc = WMMA_F16(F.w2[mt][1], b1, acc);
#pragma unroll
    for (int r = 0; r < 8; ++r) {
      int i = mt * 16 + hi * 8 + r;   // C/D layout: M = r (+8 for hi lanes)
      actB[col * 66 + i] = (_Float16)fast_tanh(acc[r] + sPb2[i]);
    }
  }
  lds_wave_fence();

  b0 = load_b_frag(actB, 0, lane);
  b1 = load_b_frag(actB, 32, lane);

  // ---- phi L3 + decoder dot: kdot = dW . (pW3 @ h2 + pb3) ----
  float kdot = 0.0f;
#pragma unroll
  for (int mt = 0; mt < 4; ++mt) {
    v8f acc = {};
    acc = WMMA_F16(F.w3[mt][0], b0, acc);
    acc = WMMA_F16(F.w3[mt][1], b1, acc);
#pragma unroll
    for (int r = 0; r < 8; ++r) {
      int i = mt * 16 + hi * 8 + r;
      kdot  = fmaf(sDW[i], acc[r] + sPb3[i], kdot);
    }
  }
  kdot += __shfl_xor(kdot, 16, 32);   // combine the two M-halves per column
  f    += __shfl_xor(f, 16, 32);
  f    += tb30;

  return fmaf(f, kdot, db0);          // dW@(K*f) + db == (dW@K)*f + db
}

__global__ __launch_bounds__(128, 1) void neural_ode_dopri5_kernel(
    const float* __restrict__ t,
    const float* __restrict__ pW1, const float* __restrict__ pb1,
    const float* __restrict__ pW2, const float* __restrict__ pb2,
    const float* __restrict__ pW3, const float* __restrict__ pb3,
    const float* __restrict__ dW,  const float* __restrict__ db,
    const float* __restrict__ tW1, const float* __restrict__ tb1,
    const float* __restrict__ tW2, const float* __restrict__ tb2,
    const float* __restrict__ tW3, const float* __restrict__ tb3,
    float* __restrict__ out, int Bn) {
  // ---- shared staging: weights as f16 (WMMA operands) / f32 (VALU layers) ----
  __shared__ _Float16 sW2[64 * 64];
  __shared__ _Float16 sW3[64 * 64];
  __shared__ _Float16 sT2[32 * 32];
  __shared__ float    sP1[128], sPb1[64], sPb2[64], sPb3[64], sDW[64];
  __shared__ float    sT1w[64], sTb1[32], sTb2[32], sT3[32];
  __shared__ _Float16 sAct[4][3][16 * 66];   // per-wave: h1 / h2 / g1 scratch

  const int tid = threadIdx.x;
  const int nth = blockDim.x;
  for (int i = tid; i < 64 * 64; i += nth) { sW2[i] = (_Float16)pW2[i]; sW3[i] = (_Float16)pW3[i]; }
  for (int i = tid; i < 32 * 32; i += nth) sT2[i] = (_Float16)tW2[i];
  for (int i = tid; i < 128;     i += nth) sP1[i] = pW1[i];
  for (int i = tid; i < 64;      i += nth) { sPb1[i] = pb1[i]; sPb2[i] = pb2[i]; sPb3[i] = pb3[i]; sDW[i] = dW[i]; }
  for (int i = tid; i < 64;      i += nth) sT1w[i] = tW1[i];
  for (int i = tid; i < 32;      i += nth) { sTb1[i] = tb1[i]; sTb2[i] = tb2[i]; sT3[i] = tW3[i]; }
  __syncthreads();

  const int wave = tid >> 5;
  const int lane = tid & 31;
  const int numTiles = (Bn + 15) >> 4;
  int tile = blockIdx.x * 4 + wave;
  if (tile >= numTiles) tile = numTiles - 1;

  // Register-resident weight fragments (loaded once).
  Frags F;
#pragma unroll
  for (int mt = 0; mt < 4; ++mt)
#pragma unroll
    for (int kt = 0; kt < 2; ++kt) {
      F.w2[mt][kt] = load_a_frag(sW2, 64, mt * 16, kt * 32, lane);
      F.w3[mt][kt] = load_a_frag(sW3, 64, mt * 16, kt * 32, lane);
    }
#pragma unroll
  for (int mt = 0; mt < 2; ++mt) F.t2[mt] = load_a_frag(sT2, 32, mt * 16, 0, lane);

  _Float16* actA = &sAct[wave][0][0];
  _Float16* actB = &sAct[wave][1][0];
  _Float16* actG = &sAct[wave][2][0];

  const int col  = lane & 15;
  int idx = tile * 16 + col;
  if (idx >= Bn) idx = Bn - 1;
  const float t1   = t[idx];
  const float db0  = db[0];
  const float tb30 = tb3[0];

  // Dopri5 coefficients
  const float C2 = 0.2f, C3 = 0.3f, C4 = 0.8f, C5 = (float)(8.0 / 9.0);
  const float A21 = 0.2f;
  const float A31 = (float)(3.0 / 40.0), A32 = (float)(9.0 / 40.0);
  const float A41 = (float)(44.0 / 45.0), A42 = (float)(-56.0 / 15.0), A43 = (float)(32.0 / 9.0);
  const float A51 = (float)(19372.0 / 6561.0), A52 = (float)(-25360.0 / 2187.0);
  const float A53 = (float)(64448.0 / 6561.0), A54 = (float)(-212.0 / 729.0);
  const float A61 = (float)(9017.0 / 3168.0), A62 = (float)(-355.0 / 33.0);
  const float A63 = (float)(46732.0 / 5247.0), A64 = (float)(49.0 / 176.0), A65 = (float)(-5103.0 / 18656.0);
  const float B1c = (float)(35.0 / 384.0), B3c = (float)(500.0 / 1113.0), B4c = (float)(125.0 / 192.0);
  const float B5c = (float)(-2187.0 / 6784.0), B6c = (float)(11.0 / 84.0);
  const float E1 = (float)(35.0 / 384.0 - 5179.0 / 57600.0);
  const float E3 = (float)(500.0 / 1113.0 - 7571.0 / 16695.0);
  const float E4 = (float)(125.0 / 192.0 - 393.0 / 640.0);
  const float E5 = (float)(-2187.0 / 6784.0 + 92097.0 / 339200.0);
  const float E6 = (float)(11.0 / 84.0 - 187.0 / 2100.0);
  const float E7 = (float)(-1.0 / 40.0);

  float tau = 0.0f, y = 0.0f, dt = 0.01f;

#define VF(TS, YS) vf_eval(t1, (TS), (YS), F, sP1, sPb1, sPb2, sPb3, sDW, \
                           sT1w, sTb1, sTb2, sT3, actA, actB, actG, db0, tb30, lane)

  for (int step = 0; step < 128; ++step) {
    float remaining = t1 - tau;
    bool  done      = remaining <= 1e-10f;
    float dt_eff    = done ? 0.0f : fminf(dt, remaining);

    float k1 = VF(tau, y);
    float k2 = VF(tau + C2 * dt_eff, fmaf(dt_eff, A21 * k1, y));
    float k3 = VF(tau + C3 * dt_eff, y + dt_eff * (A31 * k1 + A32 * k2));
    float k4 = VF(tau + C4 * dt_eff, y + dt_eff * (A41 * k1 + A42 * k2 + A43 * k3));
    float k5 = VF(tau + C5 * dt_eff, y + dt_eff * (A51 * k1 + A52 * k2 + A53 * k3 + A54 * k4));
    float k6 = VF(tau + dt_eff, y + dt_eff * (A61 * k1 + A62 * k2 + A63 * k3 + A64 * k4 + A65 * k5));
    float y5 = y + dt_eff * (B1c * k1 + B3c * k3 + B4c * k4 + B5c * k5 + B6c * k6);
    float k7 = VF(tau + dt_eff, y5);   // FSAL stage for embedded error

    float err   = dt_eff * (E1 * k1 + E3 * k3 + E4 * k4 + E5 * k5 + E6 * k6 + E7 * k7);
    float scale = 1e-6f + 1e-3f * fmaxf(fabsf(y), fabsf(y5));
    float ratio = err / scale;
    float en    = sqrtf(ratio * ratio + 1e-30f);
    bool  accept = en <= 1.0f;
    float fac = 0.9f * exp2f(-0.2f * log2f(fmaxf(en, 1e-10f)));
    fac = fminf(fmaxf(fac, 0.2f), 10.0f);

    bool take = (!done) && accept;
    tau = take ? tau + dt_eff : tau;
    y   = take ? y5 : y;
    dt  = done ? dt : fmaxf(dt_eff * fac, 1e-8f);
  }
#undef VF

  const int gidx = tile * 16 + col;
  if ((lane >> 4) == 0 && gidx < Bn) out[gidx] = y;
}

extern "C" void kernel_launch(void* const* d_in, const int* in_sizes, int n_in,
                              void* d_out, int out_size, void* d_ws, size_t ws_size,
                              hipStream_t stream) {
  (void)n_in; (void)out_size; (void)d_ws; (void)ws_size;
  const float* t_   = (const float*)d_in[0];
  const float* pW1  = (const float*)d_in[1];
  const float* pb1  = (const float*)d_in[2];
  const float* pW2  = (const float*)d_in[3];
  const float* pb2  = (const float*)d_in[4];
  const float* pW3  = (const float*)d_in[5];
  const float* pb3  = (const float*)d_in[6];
  const float* dW   = (const float*)d_in[7];
  const float* db   = (const float*)d_in[8];
  const float* tW1  = (const float*)d_in[9];
  const float* tb1  = (const float*)d_in[10];
  const float* tW2  = (const float*)d_in[11];
  const float* tb2  = (const float*)d_in[12];
  const float* tW3  = (const float*)d_in[13];
  const float* tb3  = (const float*)d_in[14];
  float* out = (float*)d_out;

  const int Bn = in_sizes[0];
  const int numTiles = (Bn + 15) / 16;     // 16 trajectories per wave (WMMA N)
  const int blocks   = (numTiles + 3) / 4; // 4 waves per block share weight LDS

  neural_ode_dopri5_kernel<<<blocks, 128, 0, stream>>>(
      t_, pW1, pb1, pW2, pb2, pW3, pb3, dW, db,
      tW1, tb1, tW2, tb2, tW3, tb3, out, Bn);
}
